// GraphAttention2_64665027609159
// MI455X (gfx1250) — matrix operand order repaired and checked by reading, not compiled, not run on hardware
//
#include <hip/hip_runtime.h>
#include <hip/hip_bf16.h>

#define N_NODES 4096
#define F_IN 256
#define F_OUT 64
#define NHEAD 4

typedef __attribute__((ext_vector_type(16))) __bf16 v16bf;
typedef __attribute__((ext_vector_type(8)))  float  v8f;

union BF16Frag { v16bf v; uint4 q[2]; unsigned short u[16]; };

// f32 -> bf16 round-to-nearest-even (returns raw bits)
__device__ __forceinline__ unsigned int f2bf(float f) {
  unsigned int u = __float_as_uint(f);
  unsigned int r = u + 0x7FFFu + ((u >> 16) & 1u);
  return r >> 16;
}

// ---------------------------------------------------------------------------
// Kernel 1: feats[h,n,o] = sum_f X[n,f] * W[h,f,o]   (bf16 WMMA, f32 accum)
// Grid: (N/64, H), 256 threads (8 waves). Each wave owns two 16x16 C tiles.
// W is staged into LDS in B-fragment-major order [kk][n][khalf][e] so each
// lane's fragment is 32 contiguous bytes -> 2x ds_load_b128, no packing.
// bf16 feats are written to global pre-swizzled into kernel-4's B layout.
// ---------------------------------------------------------------------------
__global__ void __launch_bounds__(256) gat_feats_kernel(
    const float* __restrict__ X, const float* __restrict__ W,
    float* __restrict__ feats_f32, unsigned short* __restrict__ feats_bfswz) {
  __shared__ __align__(16) unsigned short sX[64 * 256];   // [row][k], row-major
  __shared__ __align__(16) unsigned short sWs[256 * 64];  // fragment-major
  const int tid = threadIdx.x;
  const int n0 = blockIdx.x * 64;
  const int h  = blockIdx.y;

  // Stage X: packed u32 pairs, coalesced 8B global reads, b32 LDS stores
  unsigned int* sX32 = (unsigned int*)sX;
  #pragma unroll 4
  for (int e = 0; e < 32; ++e) {
    int idx2 = e * 256 + tid;                 // 8192 u32 elements
    int row = idx2 >> 7, kp = idx2 & 127;
    const float* xp = X + (size_t)(n0 + row) * F_IN + kp * 2;
    sX32[idx2] = f2bf(xp[0]) | (f2bf(xp[1]) << 16);
  }
  // Stage W swizzled: element (k,o) -> sWs[(kk*64+o)*32 + khalf*16 + e]
  const float* Wh = W + (size_t)h * F_IN * F_OUT;
  #pragma unroll 4
  for (int e = 0; e < 64; ++e) {
    int idx = e * 256 + tid;                  // coalesced global read
    int k = idx >> 6, o = idx & 63;
    sWs[(((k >> 5) * 64 + o) << 5) + (k & 16) + (k & 15)] = (unsigned short)f2bf(Wh[idx]);
  }
  __syncthreads();

  const int wave  = tid >> 5, lane = tid & 31;
  const int rsub  = wave >> 1;
  const int csub0 = (wave & 1) * 2, csub1 = csub0 + 1;
  const int lhalf = (lane >> 4) & 1, l16 = lane & 15;

  v8f c0 = {}, c1 = {};
  #pragma unroll
  for (int kk = 0; kk < 8; ++kk) {            // K = 256 in steps of 32
    BF16Frag a, b0, b1;
    // A fragment: two aligned 16B groups per lane
    const unsigned short* ap = sX + (rsub * 16 + l16) * 256 + kk * 32 + lhalf * 16;
    a.q[0] = *(const uint4*)ap;
    a.q[1] = *(const uint4*)(ap + 8);
    // B fragments: 32 contiguous bytes per lane (fragment-major layout)
    const unsigned short* bp0 = sWs + ((kk * 64 + csub0 * 16 + l16) << 5) + (lhalf << 4);
    const unsigned short* bp1 = sWs + ((kk * 64 + csub1 * 16 + l16) << 5) + (lhalf << 4);
    b0.q[0] = *(const uint4*)bp0; b0.q[1] = *(const uint4*)(bp0 + 8);
    b1.q[0] = *(const uint4*)bp1; b1.q[1] = *(const uint4*)(bp1 + 8);
    c0 = __builtin_amdgcn_wmma_f32_16x16x32_bf16(false, a.v, false, b0.v, (short)0, c0, false, false);
    c1 = __builtin_amdgcn_wmma_f32_16x16x32_bf16(false, a.v, false, b1.v, (short)0, c1, false, false);
  }

  // C/D layout: lane = column n, VGPR v = row (lanes>=16 -> row+8)
  unsigned short* fbh = feats_bfswz + (size_t)h * N_NODES * F_OUT;
  #pragma unroll
  for (int v = 0; v < 8; ++v) {
    int row  = n0 + rsub * 16 + v + lhalf * 8;
    size_t base = ((size_t)h * N_NODES + row) * F_OUT;
    int col0 = csub0 * 16 + l16, col1 = csub1 * 16 + l16;
    feats_f32[base + col0] = c0[v];
    feats_f32[base + col1] = c1[v];
    // swizzled bf16: [i-tile][f][khalf][e] with khalf=(row>>4)&1, e=row&15
    size_t sw = (size_t)(row >> 5) * 2048 + ((row >> 4) & 1) * 16 + (row & 15);
    fbh[sw + (col0 << 5)] = (unsigned short)f2bf(c0[v]);
    fbh[sw + (col1 << 5)] = (unsigned short)f2bf(c1[v]);
  }
}

// ---------------------------------------------------------------------------
// Kernel 2: s[h,n] = feats[h,n,:]·a_self[h],  t[h,n] = feats[h,n,:]·a_neigh[h]
// ---------------------------------------------------------------------------
__global__ void __launch_bounds__(256) gat_st_kernel(
    const float* __restrict__ feats_f32, const float* __restrict__ a_self,
    const float* __restrict__ a_neigh, float* __restrict__ s, float* __restrict__ t) {
  int g = blockIdx.x * 256 + threadIdx.x;     // H*N items
  int h = g >> 12;
  const float4* f4 = (const float4*)(feats_f32 + (size_t)g * F_OUT);
  const float4* as = (const float4*)(a_self  + h * F_OUT);
  const float4* an = (const float4*)(a_neigh + h * F_OUT);
  float ss = 0.f, tt = 0.f;
  #pragma unroll
  for (int o = 0; o < F_OUT / 4; ++o) {
    float4 v = f4[o], a = as[o], n = an[o];
    ss += v.x * a.x + v.y * a.y + v.z * a.z + v.w * a.w;
    tt += v.x * n.x + v.y * n.y + v.z * n.z + v.w * n.w;
  }
  s[g] = ss; t[g] = tt;
}

// ---------------------------------------------------------------------------
// Kernel 3: per-row online softmax stats for all 4 heads in one pass over A.
// One wave per row i; lanes read float4 of A (coalesced 512B/wave).
// ---------------------------------------------------------------------------
__global__ void __launch_bounds__(256) gat_stats_kernel(
    const float* __restrict__ A, const float* __restrict__ s, const float* __restrict__ t,
    float* __restrict__ rowmax, float* __restrict__ rowrcp) {
  const int wave = threadIdx.x >> 5, lane = threadIdx.x & 31;
  const int i = blockIdx.x * 8 + wave;
  float sv[NHEAD], m[NHEAD], l[NHEAD];
  #pragma unroll
  for (int h = 0; h < NHEAD; ++h) { sv[h] = s[h * N_NODES + i]; m[h] = -3.0e38f; l[h] = 0.f; }
  const float* Ar = A + (size_t)i * N_NODES;
  for (int jb = lane * 4; jb < N_NODES; jb += 128) {
    float4 av = *(const float4*)(Ar + jb);
    float mk[4] = { -1.0e10f * (0.05f - av.x), -1.0e10f * (0.05f - av.y),
                    -1.0e10f * (0.05f - av.z), -1.0e10f * (0.05f - av.w) };
    #pragma unroll
    for (int h = 0; h < NHEAD; ++h) {
      float4 tv = *(const float4*)(t + h * N_NODES + jb);
      float tx[4] = { tv.x, tv.y, tv.z, tv.w };
      #pragma unroll
      for (int q = 0; q < 4; ++q) {
        float x = sv[h] + tx[q];
        x = x > 0.f ? x : 0.2f * x;           // LeakyReLU(0.2)
        x += mk[q];                           // additive adjacency mask
        if (x > m[h]) { l[h] = l[h] * __expf(m[h] - x) + 1.0f; m[h] = x; }
        else          { l[h] += __expf(x - m[h]); }
      }
    }
  }
  #pragma unroll
  for (int h = 0; h < NHEAD; ++h) {
    #pragma unroll
    for (int off = 16; off >= 1; off >>= 1) { // wave32 butterfly merge
      float mo = __shfl_xor(m[h], off, 32);
      float lo = __shfl_xor(l[h], off, 32);
      float nm = fmaxf(m[h], mo);
      l[h] = l[h] * __expf(m[h] - nm) + lo * __expf(mo - nm);
      m[h] = nm;
    }
    if (lane == 0) { rowmax[h * N_NODES + i] = m[h]; rowrcp[h * N_NODES + i] = 1.0f / l[h]; }
  }
}

// ---------------------------------------------------------------------------
// Kernel 4: out_partial[h,j,f] = sum_i attn[h,i,j] * feats[h,i,f]
// Probabilities regenerated tile-by-tile into a double-buffered LDS tile
// (ATT_LD=40 keeps A-fragment groups 16B-aligned -> ds_load_b128).
// B fragments come straight from global pre-swizzled bf16 feats (L2-resident)
// as 2x global_load_b128 per fragment. Head 3 streams attn to d_out.
// ---------------------------------------------------------------------------
#define ATT_LD 40
__global__ void __launch_bounds__(256) gat_attn_mm_kernel(
    const float* __restrict__ A, const float* __restrict__ s, const float* __restrict__ t,
    const float* __restrict__ rowmax, const float* __restrict__ rowrcp,
    const unsigned short* __restrict__ feats_bfswz,
    float* __restrict__ partial, float* __restrict__ out_attn) {
  __shared__ __align__(16) unsigned short att[2][64 * ATT_LD]; // [j_loc][i_loc] bf16
  const int tid = threadIdx.x;
  const int jblock = blockIdx.x * 64;
  const int h = blockIdx.y;
  const int wave = tid >> 5, lane = tid & 31;
  const int lhalf = (lane >> 4) & 1, l16 = lane & 15;
  const int jsub  = wave >> 1;
  const int fsub0 = (wave & 1) * 2, fsub1 = fsub0 + 1;
  const float* s_h = s      + h * N_NODES;
  const float* t_h = t      + h * N_NODES;
  const float* m_h = rowmax + h * N_NODES;
  const float* r_h = rowrcp + h * N_NODES;
  const unsigned short* fbh = feats_bfswz + (size_t)h * N_NODES * F_OUT;

  v8f acc0 = {}, acc1 = {};

  for (int ii = 0; ii < N_NODES; ii += 32) {
    const int par = (ii >> 5) & 1;
    // Stage 64j x 32i probability tile (coalesced along j)
    #pragma unroll
    for (int e = 0; e < 8; ++e) {
      int idx = e * 256 + tid;
      int j_loc = idx & 63, i_loc = idx >> 6;
      int i_g = ii + i_loc, j_g = jblock + j_loc;
      float av = A[(size_t)i_g * N_NODES + j_g];
      float x = s_h[i_g] + t_h[j_g];
      x = x > 0.f ? x : 0.2f * x;
      x += -1.0e10f * (0.05f - av);
      float p = __expf(x - m_h[i_g]) * r_h[i_g];
      att[par][j_loc * ATT_LD + i_loc] = (unsigned short)f2bf(p);
      if (h == NHEAD - 1) out_attn[(size_t)i_g * N_NODES + j_g] = p; // mandatory output
    }
    if (ii + 32 < N_NODES)  // warm L2 for next i-tile of A
      __builtin_prefetch(&A[(size_t)(ii + 32 + (tid >> 6)) * N_NODES + jblock + (tid & 63)], 0, 0);
    __syncthreads();

    // A fragment = attn^T (16j x 32i): two aligned ds_load_b128
    BF16Frag af, b0, b1;
    const unsigned short* ap = &att[par][(jsub * 16 + l16) * ATT_LD + lhalf * 16];
    af.q[0] = *(const uint4*)ap;
    af.q[1] = *(const uint4*)(ap + 8);
    // B fragments: pre-swizzled global feats, 32 contiguous bytes per lane
    const unsigned short* fb = fbh + (size_t)(ii >> 5) * 2048 + (lhalf << 4);
    const unsigned short* bp0 = fb + ((fsub0 * 16 + l16) << 5);
    const unsigned short* bp1 = fb + ((fsub1 * 16 + l16) << 5);
    b0.q[0] = *(const uint4*)bp0; b0.q[1] = *(const uint4*)(bp0 + 8);
    b1.q[0] = *(const uint4*)bp1; b1.q[1] = *(const uint4*)(bp1 + 8);
    acc0 = __builtin_amdgcn_wmma_f32_16x16x32_bf16(false, af.v, false, b0.v, (short)0, acc0, false, false);
    acc1 = __builtin_amdgcn_wmma_f32_16x16x32_bf16(false, af.v, false, b1.v, (short)0, acc1, false, false);
    // no trailing barrier: double-buffered att tile
  }

  #pragma unroll
  for (int v = 0; v < 8; ++v) {
    int j_g = jblock + jsub * 16 + v + lhalf * 8;
    size_t base = ((size_t)h * N_NODES + j_g) * F_OUT;
    partial[base + fsub0 * 16 + l16] = acc0[v];
    partial[base + fsub1 * 16 + l16] = acc1[v];
  }
}

// ---------------------------------------------------------------------------
// Kernel 5: out[j,f] = relu( mean_h(partial[h,j,f]) + mean_h(b[h,f]) )
// ---------------------------------------------------------------------------
__global__ void __launch_bounds__(256) gat_finalize_kernel(
    const float* __restrict__ partial, const float* __restrict__ b,
    float* __restrict__ out_feat) {
  const size_t HS = (size_t)N_NODES * F_OUT;
  int g = blockIdx.x * 256 + threadIdx.x;     // 65536 threads, 4 elems each
  int o = g * 4;
  int f = o & 63;
  float4 p0 = *(const float4*)(partial + o);
  float4 p1 = *(const float4*)(partial + HS + o);
  float4 p2 = *(const float4*)(partial + 2 * HS + o);
  float4 p3 = *(const float4*)(partial + 3 * HS + o);
  float4 r;
  float mb;
  mb  = b[f + 0] + b[64 + f + 0] + b[128 + f + 0] + b[192 + f + 0];
  r.x = fmaxf(0.25f * (p0.x + p1.x + p2.x + p3.x) + 0.25f * mb, 0.f);
  mb  = b[f + 1] + b[64 + f + 1] + b[128 + f + 1] + b[192 + f + 1];
  r.y = fmaxf(0.25f * (p0.y + p1.y + p2.y + p3.y) + 0.25f * mb, 0.f);
  mb  = b[f + 2] + b[64 + f + 2] + b[128 + f + 2] + b[192 + f + 2];
  r.z = fmaxf(0.25f * (p0.z + p1.z + p2.z + p3.z) + 0.25f * mb, 0.f);
  mb  = b[f + 3] + b[64 + f + 3] + b[128 + f + 3] + b[192 + f + 3];
  r.w = fmaxf(0.25f * (p0.w + p1.w + p2.w + p3.w) + 0.25f * mb, 0.f);
  *(float4*)(out_feat + o) = r;
}

// ---------------------------------------------------------------------------
extern "C" void kernel_launch(void* const* d_in, const int* in_sizes, int n_in,
                              void* d_out, int out_size, void* d_ws, size_t ws_size,
                              hipStream_t stream) {
  (void)in_sizes; (void)n_in; (void)out_size; (void)ws_size;
  const float* X       = (const float*)d_in[0];
  const float* A       = (const float*)d_in[1];
  const float* W       = (const float*)d_in[2];
  const float* a_self  = (const float*)d_in[3];
  const float* a_neigh = (const float*)d_in[4];
  const float* b       = (const float*)d_in[5];

  float* out_feat = (float*)d_out;                 // [4096,64]
  float* out_attn = out_feat + N_NODES * F_OUT;    // [4096,4096]

  char* ws = (char*)d_ws;
  const size_t FEATS = (size_t)NHEAD * N_NODES * F_OUT;   // 1M elements
  float*          feats_f32   = (float*)ws;                             // 4 MB
  unsigned short* feats_bfswz = (unsigned short*)(ws + FEATS * 4);      // 2 MB
  char* p = ws + FEATS * 6;
  float* s      = (float*)p; p += (size_t)NHEAD * N_NODES * 4;
  float* t      = (float*)p; p += (size_t)NHEAD * N_NODES * 4;
  float* rowmax = (float*)p; p += (size_t)NHEAD * N_NODES * 4;
  float* rowrcp = (float*)p; p += (size_t)NHEAD * N_NODES * 4;
  float* partial = (float*)p;                                           // 4 MB

  gat_feats_kernel<<<dim3(N_NODES / 64, NHEAD), 256, 0, stream>>>(X, W, feats_f32, feats_bfswz);
  gat_st_kernel<<<NHEAD * N_NODES / 256, 256, 0, stream>>>(feats_f32, a_self, a_neigh, s, t);
  gat_stats_kernel<<<N_NODES / 8, 256, 0, stream>>>(A, s, t, rowmax, rowrcp);
  gat_attn_mm_kernel<<<dim3(N_NODES / 64, NHEAD), 256, 0, stream>>>(
      A, s, t, rowmax, rowrcp, feats_bfswz, partial, out_attn);
  gat_finalize_kernel<<<N_NODES * F_OUT / (256 * 4), 256, 0, stream>>>(partial, b, out_feat);
}
